// non_local_24885040513396
// MI455X (gfx1250) — matrix operand order repaired and checked by reading, not compiled, not run on hardware
//
#include <hip/hip_runtime.h>
#include <hip/hip_bf16.h>

typedef _Float16 v16h __attribute__((ext_vector_type(16)));
typedef _Float16 v8h  __attribute__((ext_vector_type(8)));
typedef _Float16 v2h  __attribute__((ext_vector_type(2)));
typedef float    v8f  __attribute__((ext_vector_type(8)));

#define NPIX (96*96)   // 9216
#define CIN  64
#define CMID 32
#define BSZ  2

union V16 { v16h v; v8h h[2]; unsigned u[8]; };

__device__ inline v16h load_A16x32(const _Float16* row, int hi) {
    // f16 A-matrix 16x32 per-lane layout: a[0..7]=K(hi*8..hi*8+7), a[8..15]=K(16+hi*8..)
    V16 t;
    t.h[0] = *(const v8h*)(row + hi * 8);
    t.h[1] = *(const v8h*)(row + 16 + hi * 8);
    return t.v;
}

__device__ inline unsigned pk2(float a, float b) {
    union { v2h h; unsigned u; } t;
    t.h.x = (_Float16)a;
    t.h.y = (_Float16)b;
    return t.u;
}

// ---------------- Kernel 1: f = relu(bn(w1 @ x)), stored f16 row-major and transposed
__global__ void conv1_bn_relu(const float* __restrict__ x, const float* __restrict__ w1,
                              const float* __restrict__ g1, const float* __restrict__ b1,
                              const float* __restrict__ m1, const float* __restrict__ v1,
                              _Float16* __restrict__ f, _Float16* __restrict__ fT) {
    int t  = blockIdx.x * blockDim.x + threadIdx.x;   // = (b*NPIX + n)*32 + c2
    int c2 = t & 31;
    int bn = t >> 5;                                  // b*NPIX + n
    int n  = bn % NPIX;
    int b  = bn / NPIX;
    const float* xp = x + ((size_t)b * CIN) * NPIX + n;
    const float* wp = w1 + c2 * CIN;
    float acc = 0.f;
#pragma unroll 8
    for (int c = 0; c < CIN; ++c) acc += wp[c] * xp[(size_t)c * NPIX];
    float inv = g1[c2] * rsqrtf(v1[c2] + 1e-5f);
    float y = (acc - m1[c2]) * inv + b1[c2];
    y = fmaxf(y, 0.f);
    _Float16 h = (_Float16)y;
    f [((size_t)bn) * CMID + c2] = h;
    fT[((size_t)b * CMID + c2) * NPIX + n] = h;
}

// ---------------- Kernel 2: flash attention, one wave per 16-query tile,
// software-pipelined: loads for k0+32 issue before softmax of k0 consumes S.
__global__ void flash_attn(const _Float16* __restrict__ f, const _Float16* __restrict__ fT,
                           float* __restrict__ y1) {
    const int lane = threadIdx.x & 31;
    const int wave = threadIdx.x >> 5;
    const int tile = blockIdx.x * (blockDim.x >> 5) + wave;   // 0..1151
    const int b  = tile / (NPIX / 16);
    const int q0 = (tile % (NPIX / 16)) * 16;
    const int hi = lane >> 4;
    const int lm = lane & 15;

    const _Float16* fb  = f  + (size_t)b * NPIX * CMID;
    const _Float16* ftb = fT + (size_t)b * CMID * NPIX;

    // Q^T as B-operand (32x16): lane = query column, elems 0-7 d=hi*8.., 8-15 d=16+hi*8..
    // Pre-scaled by log2(e): scores land in the exp2 domain (softmax invariant),
    // so every exp is a bare v_exp_f32 with no multiply.
    V16 bq;
    {
        const _Float16* qp = fb + ((size_t)(q0 + lm)) * CMID + hi * 8;
        bq.h[0] = *(const v8h*)(qp);
        bq.h[1] = *(const v8h*)(qp + 16);
        const _Float16 l2e = (_Float16)1.4426950408889634f;
#pragma unroll
        for (int i = 0; i < 16; ++i) bq.v[i] *= l2e;
    }

    v8f acc1 = {0.f,0.f,0.f,0.f,0.f,0.f,0.f,0.f};
    v8f acc2 = {0.f,0.f,0.f,0.f,0.f,0.f,0.f,0.f};
    const v8f zero = {0.f,0.f,0.f,0.f,0.f,0.f,0.f,0.f};
    float m_run = -1e30f, l_run = 0.f;

    // ---- pipeline prologue: K/V tiles and S for k0 = 0
    v8f S1, S2;
    v16h Av0, Av1;
    {
        v16h Ak0 = load_A16x32(fb + (size_t)(lm) * CMID, hi);
        v16h Ak1 = load_A16x32(fb + (size_t)(16 + lm) * CMID, hi);
        Av0 = load_A16x32(ftb + (size_t)lm        * NPIX, hi);
        Av1 = load_A16x32(ftb + (size_t)(16 + lm) * NPIX, hi);
        S1 = __builtin_amdgcn_wmma_f32_16x16x32_f16(false, Ak0, false, bq.v,
                                                    (short)0, zero, false, false);
        S2 = __builtin_amdgcn_wmma_f32_16x16x32_f16(false, Ak1, false, bq.v,
                                                    (short)0, zero, false, false);
    }

    for (int k0 = 0; k0 < NPIX; k0 += 32) {
        // ---- issue next tile's loads first (hides L2 latency + WMMA->VALU hazard)
        int kn = k0 + 32;
        if (kn >= NPIX) kn = 0;                       // harmless dummy on last iter
        v16h nAk0 = load_A16x32(fb + (size_t)(kn + lm) * CMID, hi);
        v16h nAk1 = load_A16x32(fb + (size_t)(kn + 16 + lm) * CMID, hi);
        v16h nAv0 = load_A16x32(ftb + (size_t)lm        * NPIX + kn, hi);
        v16h nAv1 = load_A16x32(ftb + (size_t)(16 + lm) * NPIX + kn, hi);

        // ---- online softmax on S (scores for keys k0 + r + 8*hi (+16), query q0+lm)
        float mx = fmaxf(S1[0], S1[1]);
#pragma unroll
        for (int r = 2; r < 8; ++r) mx = fmaxf(mx, S1[r]);
#pragma unroll
        for (int r = 0; r < 8; ++r) mx = fmaxf(mx, S2[r]);
        mx = fmaxf(mx, __shfl_xor(mx, 16, 32));
        float m_new = fmaxf(m_run, mx);
        float alpha = __builtin_amdgcn_exp2f(m_run - m_new);

        float e1[8], e2[8];
#pragma unroll
        for (int r = 0; r < 8; ++r) e1[r] = __builtin_amdgcn_exp2f(S1[r] - m_new);
#pragma unroll
        for (int r = 0; r < 8; ++r) e2[r] = __builtin_amdgcn_exp2f(S2[r] - m_new);
        // pairwise-tree sum (short dependency chains, dual-issue friendly)
        float sa = (e1[0] + e1[1]) + (e1[2] + e1[3]);
        float sb = (e1[4] + e1[5]) + (e1[6] + e1[7]);
        float sc = (e2[0] + e2[1]) + (e2[2] + e2[3]);
        float sd = (e2[4] + e2[5]) + (e2[6] + e2[7]);
        float s  = (sa + sb) + (sc + sd);
        s += __shfl_xor(s, 16, 32);
        l_run = l_run * alpha + s;
        m_run = m_new;
#pragma unroll
        for (int r = 0; r < 8; ++r) { acc1[r] *= alpha; acc2[r] *= alpha; }

        // ---- repack exp(S^T) into P B-operand layout (32 keys x 16 queries)
        // each lane packs its own e1/e2, no shuffle: S key rows 8*hi+r already
        // match the table k order (8h+s, 16+8h+s-8) used by Av0/Av1.
        V16 bp;
#pragma unroll
        for (int j = 0; j < 4; ++j) {
            unsigned p1 = pk2(e1[2*j], e1[2*j+1]);
            unsigned p2 = pk2(e2[2*j], e2[2*j+1]);
            bp.u[j]     = p1;   // keys 0-7   / 8-15
            bp.u[4 + j] = p2;   // keys 16-23 / 24-31
        }

        // ---- accumulate O^T += V^T * P
        acc1 = __builtin_amdgcn_wmma_f32_16x16x32_f16(false, Av0, false, bp.v,
                                                      (short)0, acc1, false, false);
        acc2 = __builtin_amdgcn_wmma_f32_16x16x32_f16(false, Av1, false, bp.v,
                                                      (short)0, acc2, false, false);

        // ---- next S last: max distance to its consumer (next iteration's softmax)
        S1 = __builtin_amdgcn_wmma_f32_16x16x32_f16(false, nAk0, false, bq.v,
                                                    (short)0, zero, false, false);
        S2 = __builtin_amdgcn_wmma_f32_16x16x32_f16(false, nAk1, false, bq.v,
                                                    (short)0, zero, false, false);
        Av0 = nAv0;
        Av1 = nAv1;
    }

    float inv_l = 1.0f / l_run;
    float* yb = y1 + (size_t)b * CMID * NPIX + q0 + lm;
#pragma unroll
    for (int r = 0; r < 8; ++r) {
        yb[(size_t)(r + 8 * hi) * NPIX]      = acc1[r] * inv_l;   // d = 0..15
        yb[(size_t)(16 + r + 8 * hi) * NPIX] = acc2[r] * inv_l;   // d = 16..31
    }
}

// ---------------- Kernel 3: out = x + relu(bn(w2 @ y1))
__global__ void conv2_bn_relu_add(const float* __restrict__ y1, const float* __restrict__ x,
                                  const float* __restrict__ w2,
                                  const float* __restrict__ g2, const float* __restrict__ b2,
                                  const float* __restrict__ m2, const float* __restrict__ v2,
                                  float* __restrict__ out) {
    int t  = blockIdx.x * blockDim.x + threadIdx.x;   // = (b*NPIX + n)*64 + c
    int c  = t & 63;
    int bn = t >> 6;
    int n  = bn % NPIX;
    int b  = bn / NPIX;
    const float* yp = y1 + ((size_t)b * CMID) * NPIX + n;
    const float* wp = w2 + c * CMID;
    float acc = 0.f;
#pragma unroll 8
    for (int cc = 0; cc < CMID; ++cc) acc += wp[cc] * yp[(size_t)cc * NPIX];
    float inv = g2[c] * rsqrtf(v2[c] + 1e-5f);
    float y = (acc - m2[c]) * inv + b2[c];
    y = fmaxf(y, 0.f);
    size_t idx = ((size_t)b * CIN + c) * NPIX + n;
    out[idx] = x[idx] + y;
}

extern "C" void kernel_launch(void* const* d_in, const int* in_sizes, int n_in,
                              void* d_out, int out_size, void* d_ws, size_t ws_size,
                              hipStream_t stream) {
    const float* x  = (const float*)d_in[0];
    const float* w1 = (const float*)d_in[1];
    const float* g1 = (const float*)d_in[2];
    const float* b1 = (const float*)d_in[3];
    const float* m1 = (const float*)d_in[4];
    const float* v1 = (const float*)d_in[5];
    const float* w2 = (const float*)d_in[6];
    const float* g2 = (const float*)d_in[7];
    const float* b2 = (const float*)d_in[8];
    const float* m2 = (const float*)d_in[9];
    const float* v2 = (const float*)d_in[10];
    float* out = (float*)d_out;

    // workspace layout
    char* ws = (char*)d_ws;
    _Float16* f  = (_Float16*)(ws);                                   // B*N*32 f16 = 1,179,648 B
    _Float16* fT = (_Float16*)(ws + (size_t)BSZ * NPIX * CMID * 2);   // same size
    float*    y1 = (float*)   (ws + (size_t)BSZ * NPIX * CMID * 4);   // B*32*N f32 = 2,359,296 B

    // 1) conv1 + bn + relu -> f (row-major) and fT (transposed)
    {
        int total = BSZ * NPIX * CMID;      // 589,824
        conv1_bn_relu<<<total / 256, 256, 0, stream>>>(x, w1, g1, b1, m1, v1, f, fT);
    }
    // 2) flash attention: 1152 query tiles, 8 waves/block
    {
        int tiles = BSZ * (NPIX / 16);      // 1152
        flash_attn<<<tiles / 8, 256, 0, stream>>>(f, fT, y1);
    }
    // 3) conv2 + bn + relu + residual
    {
        int total = BSZ * NPIX * CIN;       // 1,179,648
        conv2_bn_relu_add<<<total / 256, 256, 0, stream>>>(y1, x, w2, g2, b2, m2, v2, out);
    }
}
